// IGNNBlock_31044023616098
// MI455X (gfx1250) — matrix operand order repaired and checked by reading, not compiled, not run on hardware
//
#include <hip/hip_runtime.h>
#include <hip/hip_bf16.h>

typedef __attribute__((ext_vector_type(16))) _Float16 v16h;
typedef __attribute__((ext_vector_type(8)))  float    v8f;
typedef __attribute__((ext_vector_type(4)))  float    v4f;

#define N_NODES 50000
#define HID 128
#define KAPPA 0.95f
#define EPS 1e-5f
#define ROW_TILES 3125   // 50000 / 16

// ---------------------------------------------------------------- small init
__global__ void k_zero_norm(float* normAcc) { *normAcc = 0.0f; }

// out rows = bias (or zero).  grid: N*128/256 blocks
__global__ void k_rowinit(float* __restrict__ out, const float* __restrict__ bias,
                          int use_bias) {
    int idx = blockIdx.x * blockDim.x + threadIdx.x;       // 0 .. N*128-1
    if (idx >= N_NODES * HID) return;
    out[idx] = use_bias ? bias[idx & (HID - 1)] : 0.0f;
}

// ---------------------------------------------------------------- Gram: W0 = Fm^T Fm, accumulate ||W||_F^2
__global__ void k_gram(const float* __restrict__ Fm, float* __restrict__ W0,
                       float* __restrict__ normAcc) {
    int idx = blockIdx.x * blockDim.x + threadIdx.x;       // 0..16383
    int i = idx >> 7, j = idx & 127;
    float s = 0.0f;
    #pragma unroll 8
    for (int k = 0; k < HID; ++k)
        s += Fm[k * HID + i] * Fm[k * HID + j];
    W0[idx] = s;
    unsafeAtomicAdd(normAcc, s * s);
}

// ---------------------------------------------------------------- pack fp32 weight -> fp16 B-fragment layout
// element o = ((kc*8 + ct)*32 + lane)*16 + i  holds  Wsrc[K*128 + col],
//   K = kc*32 + 16*(lane/16) + i,  col = ct*16 + (lane%16)
__global__ void k_pack(const float* __restrict__ Wsrc, _Float16* __restrict__ dst,
                       const float* __restrict__ normAcc, int use_norm) {
    int o = blockIdx.x * blockDim.x + threadIdx.x;         // 0..16383
    int i    =  o        & 15;
    int lane = (o >> 4)  & 31;
    int ct   = (o >> 9)  & 7;
    int kc   =  o >> 12;
    int K   = kc * 32 + ((lane >> 4) << 4) + i;
    int col = ct * 16 + (lane & 15);
    float scale = 1.0f;
    if (use_norm) {
        float n = sqrtf(*normAcc);
        scale = (n > 1.0f) ? (1.0f / (n + EPS)) : 1.0f;
    }
    dst[o] = (_Float16)(scale * Wsrc[K * HID + col]);
}

// ---------------------------------------------------------------- WMMA GEMM: Out[tile] = alpha*f(A) @ Bpack (+ Out if LOAD_C)
// block = 256 threads = 8 waves, each wave does one 16-row strip x 128 cols.
// Compile-time specialization avoids cndmask select chains on the WMMA feed path.
template <bool LOAD_C, bool RELU, bool SCALE>
__global__ __launch_bounds__(256)
void k_gemm(const float* __restrict__ A, const _Float16* __restrict__ Bpack,
            float* __restrict__ Out, float alpha) {
    __shared__ v16h ldsB[1024];                            // 32 KB packed weights
    int tid = threadIdx.x;
    {   // cooperative stage: global -> LDS (b128 pairs)
        const v16h* gB = (const v16h*)Bpack;
        #pragma unroll
        for (int k = tid; k < 1024; k += 256) ldsB[k] = gB[k];
    }
    __syncthreads();

    int wave = tid >> 5;
    int lane = tid & 31;
    int g    = lane >> 4;          // lane half
    int m    = lane & 15;          // row within tile (A) / col (B,C,D)

    int rowTile = blockIdx.x * 8 + wave;
    if (rowTile >= ROW_TILES) return;
    int row0 = rowTile * 16;

    v8f acc[8];
    #pragma unroll
    for (int ct = 0; ct < 8; ++ct) {
        if (LOAD_C) {
            #pragma unroll
            for (int r = 0; r < 8; ++r)
                acc[ct][r] = Out[(size_t)(row0 + r + 8 * g) * HID + ct * 16 + m];
        } else {
            acc[ct] = (v8f){0.f, 0.f, 0.f, 0.f, 0.f, 0.f, 0.f, 0.f};
        }
    }

    #pragma unroll
    for (int kc = 0; kc < 4; ++kc) {
        // A fragment: contiguous k = 16g .. 16g+15 (matches packed B)
        const float* arow = A + (size_t)(row0 + m) * HID + kc * 32 + 16 * g;
        float tmp[16];
        *(v4f*)(tmp)      = *(const v4f*)(arow);
        *(v4f*)(tmp + 4)  = *(const v4f*)(arow + 4);
        *(v4f*)(tmp + 8)  = *(const v4f*)(arow + 8);
        *(v4f*)(tmp + 12) = *(const v4f*)(arow + 12);
        v16h a;
        #pragma unroll
        for (int i = 0; i < 16; ++i) {
            float v = tmp[i];
            if (RELU)  v = fmaxf(v, 0.0f);
            if (SCALE) v = alpha * v;
            a[i] = (_Float16)v;
        }
        #pragma unroll
        for (int ct = 0; ct < 8; ++ct) {
            v16h b = ldsB[(kc * 8 + ct) * 32 + lane];
            acc[ct] = __builtin_amdgcn_wmma_f32_16x16x32_f16(
                false, a, false, b, (short)0, acc[ct], false, false);
        }
    }

    #pragma unroll
    for (int ct = 0; ct < 8; ++ct)
        #pragma unroll
        for (int r = 0; r < 8; ++r)
            Out[(size_t)(row0 + r + 8 * g) * HID + ct * 16 + m] = acc[ct][r];
}

// ---------------------------------------------------------------- SPMM: out[dst] += w * in[src]; one wave per edge
__global__ __launch_bounds__(256)
void k_spmm(const float* __restrict__ in, const int* __restrict__ esrc,
            const int* __restrict__ edst, const float* __restrict__ ew,
            float* __restrict__ out, int E) {
    int e = blockIdx.x * 8 + (threadIdx.x >> 5);
    if (e >= E) return;
    int lane = threadIdx.x & 31;
    int src = esrc[e], dst = edst[e];
    float w = ew[e];
    v4f v = *(const v4f*)(in + (size_t)src * HID + lane * 4);
    float* o = out + (size_t)dst * HID + lane * 4;
    unsafeAtomicAdd(o + 0, w * v.x);
    unsafeAtomicAdd(o + 1, w * v.y);
    unsafeAtomicAdd(o + 2, w * v.z);
    unsafeAtomicAdd(o + 3, w * v.w);
}

// ---------------------------------------------------------------- launch
extern "C" void kernel_launch(void* const* d_in, const int* in_sizes, int n_in,
                              void* d_out, int out_size, void* d_ws, size_t ws_size,
                              hipStream_t stream) {
    const float* x    = (const float*)d_in[0];
    const int*   esrc = (const int*)  d_in[1];
    const int*   edst = (const int*)  d_in[2];
    const float* ew   = (const float*)d_in[3];
    const float* W1   = (const float*)d_in[4];
    const float* b1   = (const float*)d_in[5];
    const float* W2   = (const float*)d_in[6];
    const float* b2   = (const float*)d_in[7];
    const float* Fm   = (const float*)d_in[8];
    const float* emb  = (const float*)d_in[9];
    float* out = (float*)d_out;
    int E = in_sizes[1];

    char* ws = (char*)d_ws;
    float*    W0      = (float*)   (ws + 0);          // 64 KB
    float*    normAcc = (float*)   (ws + 65536);      // pad to 256
    _Float16* Wpack   = (_Float16*)(ws + 65792);      // 32 KB
    _Float16* W1pack  = (_Float16*)(ws + 98560);      // 32 KB
    _Float16* W2pack  = (_Float16*)(ws + 131328);     // 32 KB
    float*    tbuf    = (float*)   (ws + 164096);                 // N*128 f32
    float*    hbuf    = (float*)   (ws + 164096 + (size_t)N_NODES * HID * 4);

    int spmmBlocks = (E + 7) / 8;
    int gemmBlocks = (ROW_TILES + 7) / 8;

    // W = normalize(Fm^T Fm), pack weights to fp16 fragment layout
    k_zero_norm<<<1, 1, 0, stream>>>(normAcc);
    k_gram<<<64, 256, 0, stream>>>(Fm, W0, normAcc);
    k_pack<<<64, 256, 0, stream>>>(W0, Wpack, normAcc, 1);
    k_pack<<<64, 256, 0, stream>>>(W1, W1pack, normAcc, 0);
    k_pack<<<64, 256, 0, stream>>>(W2, W2pack, normAcc, 0);

    // h = spmm(x @ W1) + b1   (relu fused into next GEMM's A-load)
    k_rowinit<<<25000, 256, 0, stream>>>(hbuf, b1, 1);
    k_gemm<false, false, false><<<gemmBlocks, 256, 0, stream>>>(x, W1pack, tbuf, 1.0f);
    k_spmm<<<spmmBlocks, 256, 0, stream>>>(tbuf, esrc, edst, ew, hbuf, E);

    // base = spmm(relu(h) @ W2) + b2  -> out
    k_rowinit<<<25000, 256, 0, stream>>>(out, b2, 1);
    k_gemm<false, true, false><<<gemmBlocks, 256, 0, stream>>>(hbuf, W2pack, tbuf, 1.0f);
    k_spmm<<<spmmBlocks, 256, 0, stream>>>(tbuf, esrc, edst, ew, out, E);

    // out += kappa * (spmm(emb) @ W)
    k_rowinit<<<25000, 256, 0, stream>>>(hbuf, b1, 0);   // zero
    k_spmm<<<spmmBlocks, 256, 0, stream>>>(emb, esrc, edst, ew, hbuf, E);
    k_gemm<true, false, true><<<gemmBlocks, 256, 0, stream>>>(hbuf, Wpack, out, KAPPA);
}
